// Attention_T2T_46849503264877
// MI455X (gfx1250) — compile-verified, matrix operand order checked by reading, not run on hardware
//
#include <hip/hip_runtime.h>
#include <hip/hip_bf16.h>

// ---------------------------------------------------------------------------
// Attention block for MI455X (gfx1250, wave32, WMMA).
// D = 512, N = 4096, B = 8.  bf16 WMMA (16x16x32) with fp32 accumulation.
// TDM (tensor_load_to_lds) stages bf16 tiles into LDS; double-buffered with
// deferred s_wait_tensorcnt so DMA for step s+1 overlaps WMMAs of step s.
// ---------------------------------------------------------------------------

typedef unsigned int uint;
typedef __attribute__((ext_vector_type(16))) __bf16 v16bf;
typedef __attribute__((ext_vector_type(8)))  __bf16 v8bf;
typedef __attribute__((ext_vector_type(8)))  float  v8f;
typedef __attribute__((ext_vector_type(4)))  uint   v4u;
typedef __attribute__((ext_vector_type(8)))  int    v8i;
typedef __attribute__((ext_vector_type(4)))  int    v4i;

#define ATT_SCALE 0.044194173824159216f  // 1/sqrt(512)

#if __has_builtin(__builtin_amdgcn_tensor_load_to_lds) && __has_builtin(__builtin_amdgcn_s_wait_tensorcnt)
#define USE_TDM 1
#else
#define USE_TDM 0
#endif

// ---------------------------------------------------------------------------
// TDM: 2D bf16 tile load, tile = rows x 32 elements, contiguous in dim0.
// LDS destination gets 4 DWORDs of padding every 16 DWORDs (pad_interval=3,
// pad_amount=3) -> effective LDS row stride = 40 bf16 = 80 bytes (16B aligned).
// Descriptor packing per CDNA5 ISA ch. 8 (D# groups 0/1; groups 2/3 zero, 2D).
// 6-arg builtin: (v4u g0, v8i g1, v4i g2, v4i g3, v8i g4, i32 cpol)
// ---------------------------------------------------------------------------
#if USE_TDM
__device__ __forceinline__ void tdm_load_tile32(uint lds_off, const void* gptr,
                                                uint ld_elems, uint rows) {
  unsigned long long ga = (unsigned long long)gptr;
  v4u g0;
  g0.x = 1u;                                            // count=1, user D#
  g0.y = lds_off;                                       // LDS byte address
  g0.z = (uint)(ga & 0xffffffffull);                    // global addr [31:0]
  g0.w = (uint)((ga >> 32) & 0x01ffffffull) | (2u << 30); // addr[56:32], type=2
  v8i g1;
  uint w0 = (1u << 16)                                  // data_size = 2 bytes
          | (1u << 20)                                  // pad_enable
          | (3u << 22)                                  // pad_interval: 16 DW
          | (3u << 25);                                 // pad_amount: 4 DW
  g1[0] = (int)w0;
  g1[1] = (int)((ld_elems & 0xffffu) << 16);            // tensor_dim0[15:0]
  g1[2] = (int)((ld_elems >> 16) | ((rows & 0xffffu) << 16)); // dim0 hi, dim1 lo
  g1[3] = (int)((rows >> 16) | (32u << 16));            // dim1 hi, tile_dim0=32
  g1[4] = (int)(rows & 0xffffu);                        // tile_dim1, tile_dim2=0
  g1[5] = (int)ld_elems;                                // tensor_dim0_stride lo
  g1[6] = 0;
  g1[7] = 0;
  v4i z4 = {0, 0, 0, 0};
  v8i z8 = {0, 0, 0, 0, 0, 0, 0, 0};
  __builtin_amdgcn_tensor_load_to_lds(g0, g1, z4, z4, z8, 0);
}
#endif

// Stage a rows x 32 bf16 tile (global, row stride ld_elems) into LDS with
// row stride 40.  Issue from wave 0 only (TDM ignores EXEC; one instruction).
__device__ __forceinline__ void stage_tile(__bf16* dst40, const __bf16* src,
                                           int ld_elems, int rows) {
#if USE_TDM
  if (threadIdx.x < 32) {
    tdm_load_tile32((uint)(unsigned long long)dst40, (const void*)src,
                    (uint)ld_elems, (uint)rows);
  }
#else
  for (int i = threadIdx.x; i < rows * 2; i += (int)blockDim.x) {
    int row = i >> 1, hf = i & 1;
    v8bf* d = (v8bf*)(dst40 + row * 40 + hf * 16);
    const v8bf* s = (const v8bf*)(src + (size_t)row * ld_elems + hf * 16);
    d[0] = s[0];
    d[1] = s[1];
  }
#endif
}

// Deferred tensor wait: allow N newest TDM ops to stay in flight (the
// prefetch for the next step); everything older (current buffers) is done.
template <int N>
__device__ __forceinline__ void stage_wait() {
#if USE_TDM
  if (threadIdx.x < 32) __builtin_amdgcn_s_wait_tensorcnt((short)N);
#endif
}

// ---------------------------------------------------------------------------
// WMMA fragment helpers (CDNA5 layouts, wave32).
// A (16x32 bf16): lane l -> row = l&15; K runs {8h..8h+7, 16+8h..23+8h}, h=l>>4.
// B (32x16 bf16): identical per-lane structure with N over lanes -> source
//                 must be [n][k] row-major (contiguous K).
// C/D (16x16 f32): VGPR j -> row j + 8h, column = l&15.
// ---------------------------------------------------------------------------
__device__ __forceinline__ v16bf frag_ld(const __bf16* tile, int ld, int rc0, int k0) {
  const int lane = (int)(threadIdx.x & 31u);
  const int r = lane & 15, h = lane >> 4;
  const __bf16* p = tile + (size_t)(rc0 + r) * ld + k0 + 8 * h;
  union { v16bf v; v8bf h8[2]; } u;
  u.h8[0] = *(const v8bf*)(p);
  u.h8[1] = *(const v8bf*)(p + 16);
  return u.v;
}

__device__ __forceinline__ v8f wmma_bf16(v16bf a, v16bf b, v8f c) {
  return __builtin_amdgcn_wmma_f32_16x16x32_bf16(false, a, false, b,
                                                 (short)0, c, false, false);
}

// ---------------------------------------------------------------------------
// Kernel 1: fp32 -> bf16 weight packing (layouts kept [f][c] / [e][d] so they
// serve directly as WMMA B-matrix sources).
// ---------------------------------------------------------------------------
__global__ void k_pack(const float* __restrict__ wqkv, const float* __restrict__ wproj,
                       __bf16* __restrict__ wqkvB, __bf16* __restrict__ wprojB) {
  int i = (int)(blockIdx.x * blockDim.x + threadIdx.x);
  if (i < 1536 * 512) wqkvB[i] = (__bf16)wqkv[i];
  if (i < 512 * 512)  wprojB[i] = (__bf16)wproj[i];
}

// ---------------------------------------------------------------------------
// Kernel 2: QKV GEMM.  qkv[m,f] = sum_c x[m,c] * w_qkv[f,c].
// 128x128 tile per WG, 8 waves (4 M x 2 F), K-step 32, double-buffered:
// A tile (fp32 x -> bf16 via VALU) and B tile (TDM) for step s+1 are staged
// while step s computes; s_wait_tensorcnt<=1 retires only the current buffer.
// ---------------------------------------------------------------------------
__global__ void k_qkv(const float* __restrict__ x, const __bf16* __restrict__ wqkvB,
                      __bf16* __restrict__ qB, __bf16* __restrict__ kB,
                      __bf16* __restrict__ vT, float* __restrict__ dout) {
  __shared__ __bf16 As[2][128][40];
  __shared__ __bf16 Bs[2][128][40];
  const int mtile = (int)blockIdx.x * 128;
  const int fbase = (int)blockIdx.y * 128;
  const int w = (int)(threadIdx.x >> 5);
  const int lane = (int)(threadIdx.x & 31u);
  const int r = lane & 15, h = lane >> 4;
  const int wm = w & 3, wn = w >> 2;

  auto stageA = [&](int buf, int kk) {
    int t = (int)threadIdx.x, row = t >> 1, hf = t & 1;
    const float4* s4 = (const float4*)(x + (size_t)(mtile + row) * 512 + kk + hf * 16);
    float4 a0 = s4[0], a1 = s4[1], a2 = s4[2], a3 = s4[3];
    union { __bf16 e[16]; v8bf v[2]; } u;
    u.e[0] = (__bf16)a0.x; u.e[1] = (__bf16)a0.y; u.e[2] = (__bf16)a0.z; u.e[3] = (__bf16)a0.w;
    u.e[4] = (__bf16)a1.x; u.e[5] = (__bf16)a1.y; u.e[6] = (__bf16)a1.z; u.e[7] = (__bf16)a1.w;
    u.e[8] = (__bf16)a2.x; u.e[9] = (__bf16)a2.y; u.e[10] = (__bf16)a2.z; u.e[11] = (__bf16)a2.w;
    u.e[12] = (__bf16)a3.x; u.e[13] = (__bf16)a3.y; u.e[14] = (__bf16)a3.z; u.e[15] = (__bf16)a3.w;
    v8bf* d = (v8bf*)&As[buf][row][hf * 16];
    d[0] = u.v[0];
    d[1] = u.v[1];
  };
  auto stageB = [&](int buf, int kk) {
    stage_tile(&Bs[buf][0][0], wqkvB + (size_t)fbase * 512 + kk, 512, 128);
  };

  stageB(0, 0);
  stageA(0, 0);
  v8f acc[2][4] = {};
  for (int s = 0; s < 16; ++s) {
    const int cur = s & 1, nxt = cur ^ 1;
    if (s < 15) {
      stageB(nxt, (s + 1) * 32);
      stageA(nxt, (s + 1) * 32);
      stage_wait<1>();
    } else {
      stage_wait<0>();
    }
    __syncthreads();
    v16bf a0 = frag_ld(&As[cur][0][0], 40, wm * 32, 0);
    v16bf a1 = frag_ld(&As[cur][0][0], 40, wm * 32 + 16, 0);
#pragma unroll
    for (int nf = 0; nf < 4; ++nf) {
      v16bf bfr = frag_ld(&Bs[cur][0][0], 40, wn * 64 + nf * 16, 0);
      acc[0][nf] = wmma_bf16(a0, bfr, acc[0][nf]);
      acc[1][nf] = wmma_bf16(a1, bfr, acc[1][nf]);
    }
    __syncthreads();
  }

#pragma unroll
  for (int mf = 0; mf < 2; ++mf)
#pragma unroll
    for (int nf = 0; nf < 4; ++nf)
#pragma unroll
      for (int j = 0; j < 8; ++j) {
        int m = mtile + wm * 32 + mf * 16 + j + 8 * h;
        int fl = fbase + wn * 64 + nf * 16 + r;
        float val = acc[mf][nf][j];
        if (fbase < 512) {
          qB[(size_t)m * 512 + fl] = (__bf16)val;
        } else if (fbase < 1024) {
          kB[(size_t)m * 512 + (fl - 512)] = (__bf16)val;
        } else {
          int d = fl - 1024;
          int b = m >> 12, n = m & 4095;
          vT[((size_t)b * 512 + d) * 4096 + n] = (__bf16)val;
          dout[(size_t)m * 512 + d] = val;   // residual v in fp32
        }
      }
}

// ---------------------------------------------------------------------------
// Kernel 3: flash pass 1 — per-row running max m and sum-exp l of S = QK^T*s.
// WG = 128 query rows; wave w owns rows w*16..w*16+15.  Q,K chunks are
// double-buffered with a prefetch chain that crosses key-tile boundaries,
// so the softmax reduction overlaps the next tile's TDM.
// ---------------------------------------------------------------------------
__global__ void k_stats(const __bf16* __restrict__ qB, const __bf16* __restrict__ kB,
                        float* __restrict__ stats) {
  __shared__ __bf16 Qc[2][128][40];
  __shared__ __bf16 Kc[2][128][40];
  const int b = (int)blockIdx.y, qt = (int)blockIdx.x * 128;
  const int w = (int)(threadIdx.x >> 5);
  const int lane = (int)(threadIdx.x & 31u);
  const int r = lane & 15, h = lane >> 4;
  const __bf16* qsrc = qB + ((size_t)(b * 4096 + qt)) * 512;

  auto stageQK = [&](int buf, int kt, int c) {
    stage_tile(&Qc[buf][0][0], qsrc + c * 32, 512, 128);
    stage_tile(&Kc[buf][0][0], kB + ((size_t)(b * 4096 + kt * 128)) * 512 + c * 32, 512, 128);
  };

  float rm[8], rl[8];
#pragma unroll
  for (int j = 0; j < 8; ++j) { rm[j] = -3.0e38f; rl[j] = 0.0f; }

  stageQK(0, 0, 0);
  for (int kt = 0; kt < 32; ++kt) {
    v8f s[8] = {};
    for (int c = 0; c < 16; ++c) {
      const int cur = c & 1, nxt = cur ^ 1;
      if (c < 15) {
        stageQK(nxt, kt, c + 1);
        stage_wait<2>();
      } else if (kt < 31) {
        stageQK(nxt, kt + 1, 0);
        stage_wait<2>();
      } else {
        stage_wait<0>();
      }
      __syncthreads();
      v16bf a = frag_ld(&Qc[cur][0][0], 40, w * 16, 0);
#pragma unroll
      for (int f = 0; f < 8; ++f)
        s[f] = wmma_bf16(a, frag_ld(&Kc[cur][0][0], 40, f * 16, 0), s[f]);
      __syncthreads();
    }
#pragma unroll
    for (int f = 0; f < 8; ++f) s[f] = s[f] * ATT_SCALE;
#pragma unroll
    for (int j = 0; j < 8; ++j) {
      float mv = s[0][j];
#pragma unroll
      for (int f = 1; f < 8; ++f) mv = fmaxf(mv, s[f][j]);
      for (int d2 = 1; d2 < 16; d2 <<= 1) mv = fmaxf(mv, __shfl_xor(mv, d2, 32));
      float mn = fmaxf(rm[j], mv);
      float sm = 0.0f;
#pragma unroll
      for (int f = 0; f < 8; ++f) sm += __expf(s[f][j] - mn);
      for (int d2 = 1; d2 < 16; d2 <<= 1) sm += __shfl_xor(sm, d2, 32);
      rl[j] = rl[j] * __expf(rm[j] - mn) + sm;
      rm[j] = mn;
    }
  }
  if (r == 0) {
#pragma unroll
    for (int j = 0; j < 8; ++j) {
      int row = b * 4096 + qt + w * 16 + j + 8 * h;
      stats[2 * row] = rm[j];
      stats[2 * row + 1] = rl[j];
    }
  }
}

// ---------------------------------------------------------------------------
// Kernel 4: flash pass 2 — O = softmax(S) @ V with known (m, l): no rescale.
// WG = 64 query rows x 128 d-columns; 8 waves = 4 (q) x 2 (d).
// S-phase (Q,K) and V-phase buffers are live in disjoint phases and alias the
// same LDS region; both phases double-buffer their TDM loads.
// ---------------------------------------------------------------------------
__global__ void k_pv(const __bf16* __restrict__ qB, const __bf16* __restrict__ kB,
                     const __bf16* __restrict__ vT, const float* __restrict__ stats,
                     float* __restrict__ oF) {
  __shared__ __align__(16) char smem[30720];     // S: Qc[2][64][40]+Kc[2][128][40]
  __shared__ __bf16 Pl[4][16][136];              // V: Vc[2][128][40] aliases smem
  __bf16* QcB = (__bf16*)smem;                   // 2 * 64*40  = 10240 B
  __bf16* KcB = (__bf16*)(smem + 10240);         // 2 * 128*40 = 20480 B
  __bf16* VcB = (__bf16*)smem;                   // 2 * 128*40 = 20480 B

  const int qt = (int)blockIdx.x * 64;
  const int dt = (int)blockIdx.y * 128;
  const int b = (int)blockIdx.z;
  const int w = (int)(threadIdx.x >> 5);
  const int wq = w & 3, wd = w >> 2;
  const int lane = (int)(threadIdx.x & 31u);
  const int r = lane & 15, h = lane >> 4;

  auto stageQ = [&](int buf, int c) {
    stage_tile(QcB + buf * (64 * 40), qB + ((size_t)(b * 4096 + qt)) * 512 + c * 32, 512, 64);
  };
  auto stageK = [&](int buf, int kt, int c) {
    stage_tile(KcB + buf * (128 * 40), kB + ((size_t)(b * 4096 + kt * 128)) * 512 + c * 32, 512, 128);
  };
  auto stageV = [&](int buf, int kt, int c2) {
    stage_tile(VcB + buf * (128 * 40), vT + ((size_t)b * 512 + dt) * 4096 + kt * 128 + c2 * 32, 4096, 128);
  };

  float mj[8], ilj[8];
#pragma unroll
  for (int j = 0; j < 8; ++j) {
    int row = b * 4096 + qt + wq * 16 + j + 8 * h;
    mj[j] = stats[2 * row];
    ilj[j] = 1.0f / stats[2 * row + 1];
  }

  v8f oacc[4] = {};
  for (int kt = 0; kt < 32; ++kt) {
    // ---- S = Q K^T for this key tile (double-buffered Q,K chunks) ----
    v8f s[8] = {};
    stageQ(0, 0);
    stageK(0, kt, 0);
    for (int c = 0; c < 16; ++c) {
      const int cur = c & 1, nxt = cur ^ 1;
      if (c < 15) {
        stageQ(nxt, c + 1);
        stageK(nxt, kt, c + 1);
        stage_wait<2>();
      } else {
        stage_wait<0>();
      }
      __syncthreads();
      v16bf a = frag_ld(QcB + cur * (64 * 40), 40, wq * 16, 0);
#pragma unroll
      for (int f = 0; f < 8; ++f)
        s[f] = wmma_bf16(a, frag_ld(KcB + cur * (128 * 40), 40, f * 16, 0), s[f]);
      __syncthreads();
    }
    // ---- P = exp(S*scale - m) / l  (bf16, re-fragmented through LDS) ----
    if (wd == 0) {
#pragma unroll
      for (int f = 0; f < 8; ++f)
#pragma unroll
        for (int j = 0; j < 8; ++j) {
          float pv = __expf(s[f][j] * ATT_SCALE - mj[j]) * ilj[j];
          Pl[wq][j + 8 * h][f * 16 + r] = (__bf16)pv;
        }
    }
    __syncthreads();
    // ---- O += P @ V (V^T tiles, double-buffered) ----
    stageV(0, kt, 0);
    for (int c2 = 0; c2 < 4; ++c2) {
      const int cur = c2 & 1, nxt = cur ^ 1;
      if (c2 < 3) {
        stageV(nxt, kt, c2 + 1);
        stage_wait<1>();
      } else {
        stage_wait<0>();
      }
      __syncthreads();
      v16bf a = frag_ld(&Pl[wq][0][0], 136, 0, c2 * 32);
#pragma unroll
      for (int nf = 0; nf < 4; ++nf)
        oacc[nf] = wmma_bf16(a, frag_ld(VcB + cur * (128 * 40), 40, wd * 64 + nf * 16, 0), oacc[nf]);
      __syncthreads();
    }
  }

#pragma unroll
  for (int nf = 0; nf < 4; ++nf)
#pragma unroll
    for (int j = 0; j < 8; ++j) {
      int row = b * 4096 + qt + wq * 16 + j + 8 * h;
      int col = dt + wd * 64 + nf * 16 + r;
      oF[(size_t)row * 512 + col] = oacc[nf][j];
    }
}

// ---------------------------------------------------------------------------
// Kernel 5: out = o @ w_proj^T + b_proj + v   (v already resident in d_out).
// Same pipelined tiling as k_qkv; A from fp32 o, B tile TDM from wprojB.
// ---------------------------------------------------------------------------
__global__ void k_proj(const float* __restrict__ oF, const __bf16* __restrict__ wprojB,
                       const float* __restrict__ bp, float* __restrict__ out) {
  __shared__ __bf16 As[2][128][40];
  __shared__ __bf16 Bs[2][128][40];
  const int mtile = (int)blockIdx.x * 128;
  const int fbase = (int)blockIdx.y * 128;
  const int w = (int)(threadIdx.x >> 5);
  const int lane = (int)(threadIdx.x & 31u);
  const int r = lane & 15, h = lane >> 4;
  const int wm = w & 3, wn = w >> 2;

  auto stageA = [&](int buf, int kk) {
    int t = (int)threadIdx.x, row = t >> 1, hf = t & 1;
    const float4* s4 = (const float4*)(oF + (size_t)(mtile + row) * 512 + kk + hf * 16);
    float4 a0 = s4[0], a1 = s4[1], a2 = s4[2], a3 = s4[3];
    union { __bf16 e[16]; v8bf v[2]; } u;
    u.e[0] = (__bf16)a0.x; u.e[1] = (__bf16)a0.y; u.e[2] = (__bf16)a0.z; u.e[3] = (__bf16)a0.w;
    u.e[4] = (__bf16)a1.x; u.e[5] = (__bf16)a1.y; u.e[6] = (__bf16)a1.z; u.e[7] = (__bf16)a1.w;
    u.e[8] = (__bf16)a2.x; u.e[9] = (__bf16)a2.y; u.e[10] = (__bf16)a2.z; u.e[11] = (__bf16)a2.w;
    u.e[12] = (__bf16)a3.x; u.e[13] = (__bf16)a3.y; u.e[14] = (__bf16)a3.z; u.e[15] = (__bf16)a3.w;
    v8bf* d = (v8bf*)&As[buf][row][hf * 16];
    d[0] = u.v[0];
    d[1] = u.v[1];
  };
  auto stageB = [&](int buf, int kk) {
    stage_tile(&Bs[buf][0][0], wprojB + (size_t)fbase * 512 + kk, 512, 128);
  };

  stageB(0, 0);
  stageA(0, 0);
  v8f acc[2][4] = {};
  for (int s = 0; s < 16; ++s) {
    const int cur = s & 1, nxt = cur ^ 1;
    if (s < 15) {
      stageB(nxt, (s + 1) * 32);
      stageA(nxt, (s + 1) * 32);
      stage_wait<1>();
    } else {
      stage_wait<0>();
    }
    __syncthreads();
    v16bf a0 = frag_ld(&As[cur][0][0], 40, wm * 32, 0);
    v16bf a1 = frag_ld(&As[cur][0][0], 40, wm * 32 + 16, 0);
#pragma unroll
    for (int nf = 0; nf < 4; ++nf) {
      v16bf bfr = frag_ld(&Bs[cur][0][0], 40, wn * 64 + nf * 16, 0);
      acc[0][nf] = wmma_bf16(a0, bfr, acc[0][nf]);
      acc[1][nf] = wmma_bf16(a1, bfr, acc[1][nf]);
    }
    __syncthreads();
  }

#pragma unroll
  for (int mf = 0; mf < 2; ++mf)
#pragma unroll
    for (int nf = 0; nf < 4; ++nf)
#pragma unroll
      for (int j = 0; j < 8; ++j) {
        int m = mtile + wm * 32 + mf * 16 + j + 8 * h;
        int f = fbase + wn * 64 + nf * 16 + r;
        size_t idx = (size_t)m * 512 + f;
        out[idx] = acc[mf][nf][j] + bp[f] + out[idx];  // + residual v
      }
}

// ---------------------------------------------------------------------------
// Host launcher.  Workspace layout (~162.3 MB):
//   qB  bf16 [8*4096][512]   32 MB
//   kB  bf16 [8*4096][512]   32 MB
//   vT  bf16 [8][512][4096]  32 MB
//   oF  f32  [8*4096][512]   64 MB
//   wqkvB bf16 [1536][512]  1.5 MB
//   wprojB bf16 [512][512]  0.5 MB
//   stats f32 [8*4096][2]  256 KB
// ---------------------------------------------------------------------------
extern "C" void kernel_launch(void* const* d_in, const int* in_sizes, int n_in,
                              void* d_out, int out_size, void* d_ws, size_t ws_size,
                              hipStream_t stream) {
  (void)in_sizes; (void)n_in; (void)out_size; (void)ws_size;
  const float* x     = (const float*)d_in[0];
  const float* wqkv  = (const float*)d_in[1];
  const float* wproj = (const float*)d_in[2];
  const float* bp    = (const float*)d_in[3];
  float* out = (float*)d_out;

  const size_t MD = (size_t)32768 * 512;
  __bf16* qB     = (__bf16*)d_ws;
  __bf16* kB     = qB + MD;
  __bf16* vT     = kB + MD;
  float*  oF     = (float*)(vT + MD);
  __bf16* wqkvB  = (__bf16*)(oF + MD);
  __bf16* wprojB = wqkvB + (size_t)1536 * 512;
  float*  stats  = (float*)(wprojB + (size_t)512 * 512);

  k_pack <<<3072, 256, 0, stream>>>(wqkv, wproj, wqkvB, wprojB);
  k_qkv  <<<dim3(256, 12), 256, 0, stream>>>(x, wqkvB, qB, kB, vT, out);
  k_stats<<<dim3(32, 8),   256, 0, stream>>>(qB, kB, stats);
  k_pv   <<<dim3(64, 4, 8),256, 0, stream>>>(qB, kB, vT, stats, oF);
  k_proj <<<dim3(256, 4),  256, 0, stream>>>(oF, wprojB, bp, out);
}
